// CricketHeteroGNNWithPooling_75814762709601
// MI455X (gfx1250) — matrix operand order repaired and checked by reading, not compiled, not run on hardware
//
#include <hip/hip_runtime.h>
#include <math.h>

typedef __attribute__((ext_vector_type(16))) _Float16 v16h;
typedef __attribute__((ext_vector_type(8)))  float    v8f;

#define H 128
#define SBUF 2048   // max segment length held in LDS (mean is 128; safety cap)

// ---------------------------------------------------------------------------
// Kernel B: segment start offsets via binary search (ball_batch is sorted)
// ---------------------------------------------------------------------------
__global__ void segstart_kernel(const int* __restrict__ bb, int* __restrict__ seg,
                                int N, int B)
{
    int b = blockIdx.x * blockDim.x + threadIdx.x;
    if (b > B) return;
    if (b == B) { seg[B] = N; return; }
    int lo = 0, hi = N;
    while (lo < hi) {
        int mid = (lo + hi) >> 1;
        if (bb[mid] < b) lo = mid + 1; else hi = mid;
    }
    seg[b] = lo;
}

// ---------------------------------------------------------------------------
// Fused kernel: per-segment attention-score MLP (WMMA f16->f32), segment
// softmax, and weighted-sum pooling. One 8-wave block per segment; ball rows
// are read twice but the second read hits L2 (per-block working set ~64KB).
// ---------------------------------------------------------------------------
__global__ __launch_bounds__(256) void fused_attn_pool_kernel(
    const float* __restrict__ ball,
    const float* __restrict__ attn_w1,   // [128,64] row-major
    const float* __restrict__ attn_b1,   // [64]
    const float* __restrict__ attn_w2,   // [64]
    const float* __restrict__ attn_b2,   // [1]
    const int*   __restrict__ seg,
    float* __restrict__ pooled, int N)
{
    // Packed W1 in WMMA B-matrix lane layout: [ctile][kstep][lane][8 dwords]
    __shared__ alignas(32) unsigned int w1pack[4][4][32][8];   // 16 KB
    __shared__ float sbuf[SBUF];                               // scores -> weights
    __shared__ float red[256];

    const int b    = blockIdx.x;
    const int tid  = threadIdx.x;
    const int wave = tid >> 5;
    const int lane = tid & 31;
    const int hi   = lane >> 4;     // half-wave selector
    const int ln16 = lane & 15;

    const int s0 = seg[b];
    const int s1 = seg[b + 1];
    int len = s1 - s0;
    if (len > SBUF) len = SBUF;     // memory-safety cap (statistically unreachable)

    // ---- cooperative pack of W1 (f32 -> f16 pairs) into LDS ----
    // B 32x16 f16 layout: lane l holds column N=l%16; e0..7 -> K = e + 8*(l/16), e8..15 -> K = 8 + e + 8*(l/16)
    for (int i = tid; i < 4096; i += 256) {
        int dword = i & 7;
        int l     = (i >> 3) & 31;
        int ks    = (i >> 8) & 3;
        int ct    = (i >> 10) & 3;
        int col   = ct * 16 + (l & 15);
        int kbase = ks * 32 + 2 * dword + 8 * (l >> 4) + 8 * (dword >> 2);
        union { _Float16 h[2]; unsigned int u; } u;
        u.h[0] = (_Float16)attn_w1[(kbase + 0) * 64 + col];
        u.h[1] = (_Float16)attn_w1[(kbase + 1) * 64 + col];
        w1pack[ct][ks][l][dword] = u.u;
    }

    const float bias2 = attn_b2[0];
    // per-lane second-layer constants: this lane's hidden columns ct*16+ln16
    float b1v[4], w2v[4];
    #pragma unroll
    for (int ct = 0; ct < 4; ++ct) {
        int col = ct * 16 + ln16;
        b1v[ct] = attn_b1[col];
        w2v[ct] = attn_w2[col];
    }
    __syncthreads();

    // ================= scoring pass (WMMA) =================
    if (len > 0) {
        const int len16 = (len + 15) & ~15;
        for (int base = 0; base < len16; base += 128) {
            const int tr = base + wave * 16;       // wave-uniform tile origin
            if (tr < len16) {
                int r = tr + ln16;                 // A-matrix row for this lane
                if (r >= len) r = len - 1;         // clamp: keep EXEC all-ones
                const float* rowp = ball + (size_t)(s0 + r) * H;

                v8f acc[4];
                #pragma unroll
                for (int ct = 0; ct < 4; ++ct) acc[ct] = (v8f){0,0,0,0,0,0,0,0};

                #pragma unroll
                for (int ks = 0; ks < 4; ++ks) {
                    // A 16x32 f16 layout: lanes 0-15: e0..7->K0..7, e8..15->K16..23 (hi: +8)
                    const float* rp = rowp + ks * 32 + hi * 8;
                    float4 f0 = *(const float4*)(rp);
                    float4 f1 = *(const float4*)(rp + 4);
                    float4 f2 = *(const float4*)(rp + 16);
                    float4 f3 = *(const float4*)(rp + 20);
                    v16h a;
                    a[0]=(_Float16)f0.x;  a[1]=(_Float16)f0.y;  a[2]=(_Float16)f0.z;  a[3]=(_Float16)f0.w;
                    a[4]=(_Float16)f1.x;  a[5]=(_Float16)f1.y;  a[6]=(_Float16)f1.z;  a[7]=(_Float16)f1.w;
                    a[8]=(_Float16)f2.x;  a[9]=(_Float16)f2.y;  a[10]=(_Float16)f2.z; a[11]=(_Float16)f2.w;
                    a[12]=(_Float16)f3.x; a[13]=(_Float16)f3.y; a[14]=(_Float16)f3.z; a[15]=(_Float16)f3.w;

                    #pragma unroll
                    for (int ct = 0; ct < 4; ++ct) {
                        v16h bm = *(const v16h*)(&w1pack[ct][ks][lane][0]);
                        acc[ct] = __builtin_amdgcn_wmma_f32_16x16x32_f16(
                            false, a, false, bm, (short)0, acc[ct], false, false);
                    }
                }

                // layer 2 (64 -> 1): tanh + dot with w2, reduced across the 16
                // N-lanes of each half-wave via shfl_xor (bits 0..3 only).
                // D layout: lanes 0-15 hold rows v (v=0..7), lanes 16-31 rows v+8.
                float p[8];
                #pragma unroll
                for (int v = 0; v < 8; ++v) {
                    float pv = 0.f;
                    #pragma unroll
                    for (int ct = 0; ct < 4; ++ct)
                        pv += tanhf(acc[ct][v] + b1v[ct]) * w2v[ct];
                    #pragma unroll
                    for (int off = 8; off > 0; off >>= 1)
                        pv += __shfl_xor(pv, off);
                    p[v] = pv;
                }
                if (ln16 < 8) {
                    float mine = p[0];
                    if (ln16 == 1) mine = p[1];
                    if (ln16 == 2) mine = p[2];
                    if (ln16 == 3) mine = p[3];
                    if (ln16 == 4) mine = p[4];
                    if (ln16 == 5) mine = p[5];
                    if (ln16 == 6) mine = p[6];
                    if (ln16 == 7) mine = p[7];
                    int rr = tr + 8 * hi + ln16;
                    if (rr < len) sbuf[rr] = mine + bias2;
                }
            }
        }
    }
    __syncthreads();

    // ================= segment softmax (in LDS) =================
    float mx = -INFINITY;
    for (int n = tid; n < len; n += 256) mx = fmaxf(mx, sbuf[n]);
    red[tid] = mx; __syncthreads();
    for (int off = 128; off > 0; off >>= 1) {
        if (tid < off) red[tid] = fmaxf(red[tid], red[tid + off]);
        __syncthreads();
    }
    mx = red[0]; __syncthreads();

    float sm = 0.f;
    for (int n = tid; n < len; n += 256) sm += expf(sbuf[n] - mx);
    red[tid] = sm; __syncthreads();
    for (int off = 128; off > 0; off >>= 1) {
        if (tid < off) red[tid] += red[tid + off];
        __syncthreads();
    }
    float denom = red[0]; __syncthreads();
    float inv = (denom > 0.f) ? (1.0f / denom) : 0.f;

    for (int n = tid; n < len; n += 256) sbuf[n] = expf(sbuf[n] - mx) * inv;
    __syncthreads();

    // ================= weighted-sum pooling (L2-resident re-read) ==========
    const int c = tid & 127;        // channel
    const int g = tid >> 7;         // node-interleave group (0/1)
    float acc = 0.f;
    for (int n = g; n < len; n += 2) {
        if (n + 8 < len)
            __builtin_prefetch(ball + (size_t)(s0 + n + 8) * H + c, 0, 1);
        acc += sbuf[n] * ball[(size_t)(s0 + n) * H + c];
    }
    red[tid] = acc; __syncthreads();
    if (tid < 128)
        pooled[(size_t)b * H + tid] = red[tid] + red[tid + 128];
}

// ---------------------------------------------------------------------------
// Kernel D: combiner Linear(256->128) + LayerNorm + exact GELU + two heads.
// One block (128 threads) per batch row.
// ---------------------------------------------------------------------------
__global__ __launch_bounds__(128) void head_kernel(
    const float* __restrict__ query, const float* __restrict__ pooled,
    const float* __restrict__ comb_w, const float* __restrict__ comb_b,
    const float* __restrict__ ln_g,   const float* __restrict__ ln_b,
    const float* __restrict__ bnd_w1, const float* __restrict__ bnd_b1,
    const float* __restrict__ bnd_w2, const float* __restrict__ bnd_b2,
    const float* __restrict__ wkt_w1, const float* __restrict__ wkt_b1,
    const float* __restrict__ wkt_w2, const float* __restrict__ wkt_b2,
    float* __restrict__ out, int B)
{
    const int b = blockIdx.x;
    const int tid = threadIdx.x;
    __shared__ float x[256];
    __shared__ float hh[128];
    __shared__ float red[128];

    x[tid]       = query [(size_t)b * H + tid];
    x[128 + tid] = pooled[(size_t)b * H + tid];
    __syncthreads();

    // combiner: h[tid] = x . comb_w[:, tid]  (coalesced across threads per k)
    float acc = comb_b[tid];
    #pragma unroll 8
    for (int k = 0; k < 256; ++k) acc += x[k] * comb_w[k * H + tid];

    // LayerNorm
    red[tid] = acc; __syncthreads();
    for (int off = 64; off > 0; off >>= 1) {
        if (tid < off) red[tid] += red[tid + off];
        __syncthreads();
    }
    float mu = red[0] * (1.0f / H); __syncthreads();
    float d = acc - mu;
    red[tid] = d * d; __syncthreads();
    for (int off = 64; off > 0; off >>= 1) {
        if (tid < off) red[tid] += red[tid + off];
        __syncthreads();
    }
    float var = red[0] * (1.0f / H); __syncthreads();
    float hn = d * rsqrtf(var + 1e-5f) * ln_g[tid] + ln_b[tid];

    // exact GELU
    float gelu = 0.5f * hn * (1.0f + erff(hn * 0.70710678118654752440f));
    hh[tid] = gelu; __syncthreads();

    // heads: Linear(128->64) -> ReLU -> Linear(64->1)
    float gb = 0.f, gw = 0.f;
    if (tid < 64) {
        float a1 = bnd_b1[tid], a2 = wkt_b1[tid];
        #pragma unroll 4
        for (int k = 0; k < 128; ++k) {
            float hv = hh[k];
            a1 += hv * bnd_w1[k * 64 + tid];
            a2 += hv * wkt_w1[k * 64 + tid];
        }
        a1 = fmaxf(a1, 0.f); a2 = fmaxf(a2, 0.f);
        gb = a1 * bnd_w2[tid];
        gw = a2 * wkt_w2[tid];
    }
    red[tid] = gb; __syncthreads();
    for (int off = 64; off > 0; off >>= 1) {
        if (tid < off) red[tid] += red[tid + off];
        __syncthreads();
    }
    float boundary = red[0] + bnd_b2[0]; __syncthreads();
    red[tid] = gw; __syncthreads();
    for (int off = 64; off > 0; off >>= 1) {
        if (tid < off) red[tid] += red[tid + off];
        __syncthreads();
    }
    if (tid == 0) {
        out[b]     = boundary;
        out[B + b] = red[0] + wkt_b2[0];
    }
}

// ---------------------------------------------------------------------------
extern "C" void kernel_launch(void* const* d_in, const int* in_sizes, int n_in,
                              void* d_out, int out_size, void* d_ws, size_t ws_size,
                              hipStream_t stream)
{
    const float* query     = (const float*)d_in[0];
    const float* ball      = (const float*)d_in[1];
    const int*   ball_batch= (const int*)  d_in[2];
    const float* attn_w1   = (const float*)d_in[3];
    const float* attn_b1   = (const float*)d_in[4];
    const float* attn_w2   = (const float*)d_in[5];
    const float* attn_b2   = (const float*)d_in[6];
    const float* comb_w    = (const float*)d_in[7];
    const float* comb_b    = (const float*)d_in[8];
    const float* ln_g      = (const float*)d_in[9];
    const float* ln_b      = (const float*)d_in[10];
    const float* bnd_w1    = (const float*)d_in[11];
    const float* bnd_b1    = (const float*)d_in[12];
    const float* bnd_w2    = (const float*)d_in[13];
    const float* bnd_b2    = (const float*)d_in[14];
    const float* wkt_w1    = (const float*)d_in[15];
    const float* wkt_b1    = (const float*)d_in[16];
    const float* wkt_w2    = (const float*)d_in[17];
    const float* wkt_b2    = (const float*)d_in[18];

    const int N = in_sizes[2];          // ball_batch length
    const int B = in_sizes[0] / H;      // query rows

    // workspace: seg_start[B+1] | pooled[B*H]
    char* wsb = (char*)d_ws;
    int* seg = (int*)wsb;
    size_t off = ((size_t)(B + 1) * 4 + 255) & ~(size_t)255;
    float* pooled = (float*)(wsb + off);

    float* out = (float*)d_out;

    segstart_kernel<<<dim3((B + 1 + 255) / 256), dim3(256), 0, stream>>>(
        ball_batch, seg, N, B);

    fused_attn_pool_kernel<<<dim3(B), dim3(256), 0, stream>>>(
        ball, attn_w1, attn_b1, attn_w2, attn_b2, seg, pooled, N);

    head_kernel<<<dim3(B), dim3(128), 0, stream>>>(
        query, pooled, comb_w, comb_b, ln_g, ln_b,
        bnd_w1, bnd_b1, bnd_w2, bnd_b2,
        wkt_w1, wkt_b1, wkt_w2, wkt_b2, out, B);
}
